// SSAccuracyLoss1_46325517254718
// MI455X (gfx1250) — compile-verified, matrix operand order checked by reading, not run on hardware
//
#include <hip/hip_runtime.h>

typedef __attribute__((ext_vector_type(16))) _Float16 v16h;
typedef __attribute__((ext_vector_type(8)))  float    v8f;
typedef __attribute__((ext_vector_type(4)))  float    v4f;

#define ACC_THREADS 256
#define ACC_BLOCKS  2048   // 524288 lanes -> 8 positions/lane for N = 4.19M (f16-exact counts)

__global__ void SSAccLoss_zero_ws(float* ws) {
    if (threadIdx.x < 2) ws[threadIdx.x] = 0.0f;
}

__global__ __launch_bounds__(ACC_THREADS)
void SSAccLoss_accum(const float* __restrict__ real,
                     const float* __restrict__ pred,
                     float* __restrict__ ws, int N) {
    const v4f* __restrict__ real4 = (const v4f*)real;
    const v4f* __restrict__ pred4 = (const v4f*)pred;

    const int gtid   = blockIdx.x * blockDim.x + threadIdx.x;
    const int stride = gridDim.x * blockDim.x;

    int corr = 0, tot = 0;

    for (int p = gtid; p < N; p += stride) {
        // 4x global_load_b128, non-temporal (streamed once; don't pollute L2)
        v4f p0 = __builtin_nontemporal_load(&pred4[2 * p]);
        v4f p1 = __builtin_nontemporal_load(&pred4[2 * p + 1]);
        v4f r0 = __builtin_nontemporal_load(&real4[2 * p]);
        v4f r1 = __builtin_nontemporal_load(&real4[2 * p + 1]);

        // argmax over 8 classes, first-index tie-break (strict >)
        float m = p0.x; int idx = 0;
        if (p0.y > m) { m = p0.y; idx = 1; }
        if (p0.z > m) { m = p0.z; idx = 2; }
        if (p0.w > m) { m = p0.w; idx = 3; }
        if (p1.x > m) { m = p1.x; idx = 4; }
        if (p1.y > m) { m = p1.y; idx = 5; }
        if (p1.z > m) { m = p1.z; idx = 6; }
        if (p1.w > m) { m = p1.w; idx = 7; }

        v4f rs = (idx & 4) ? r1 : r0;
        const int j = idx & 3;
        float picked = (j == 0) ? rs.x : (j == 1) ? rs.y : (j == 2) ? rs.z : rs.w;

        float rsum = (r0.x + r0.y + r0.z + r0.w) + (r1.x + r1.y + r1.z + r1.w);
        bool valid = (rsum != 0.0f);

        tot  += valid ? 1 : 0;
        corr += (valid && (picked > 0.0f)) ? 1 : 0;
    }

    // ---- wave reduction stage 1: ones-matrix WMMA merges lane n with lane n+16 ----
    // D[m,n] = sum_k A[m,k]*B[k,n]; A = ones(16x32 f16), B has one count in slot 0
    // per lane (column n = lane%16, k-half selected by lane/16). Counts <= 8, f16-exact.
    v16h a;
#pragma unroll
    for (int i = 0; i < 16; ++i) a[i] = (_Float16)1.0f;
    v16h bc = {}; bc[0] = (_Float16)(float)corr;
    v16h bt = {}; bt[0] = (_Float16)(float)tot;
    v8f  z  = {};

    v8f dc = __builtin_amdgcn_wmma_f32_16x16x32_f16(false, a, false, bc, (short)0, z, false, false);
    v8f dt = __builtin_amdgcn_wmma_f32_16x16x32_f16(false, a, false, bt, (short)0, z, false, false);

    float sc = dc[0];   // = corr[n] + corr[n+16], n = lane%16 (replicated in both half-waves)
    float st = dt[0];

    // ---- stage 2: fold the 16 columns (masks < 16 stay within each half-wave) ----
#pragma unroll
    for (int mask = 8; mask >= 1; mask >>= 1) {
        sc += __shfl_xor(sc, mask, 32);
        st += __shfl_xor(st, mask, 32);
    }

    __shared__ float s_c, s_t;
    if (threadIdx.x == 0) { s_c = 0.0f; s_t = 0.0f; }
    __syncthreads();
    if ((threadIdx.x & 31) == 0) {
        atomicAdd(&s_c, sc);
        atomicAdd(&s_t, st);
    }
    __syncthreads();
    if (threadIdx.x == 0) {
        atomicAdd(&ws[0], s_c);   // exact: integer-valued fp32 sums < 2^24
        atomicAdd(&ws[1], s_t);
    }
}

__global__ void SSAccLoss_finish(const float* __restrict__ ws, float* __restrict__ out) {
    if (threadIdx.x == 0) out[0] = 1.0f - ws[0] / ws[1];
}

extern "C" void kernel_launch(void* const* d_in, const int* in_sizes, int n_in,
                              void* d_out, int out_size, void* d_ws, size_t ws_size,
                              hipStream_t stream) {
    const float* real = (const float*)d_in[0];   // [B, L, C] fp32
    const float* pred = (const float*)d_in[1];   // [B, L, C] fp32
    float* ws  = (float*)d_ws;                   // ws[0]=correct, ws[1]=total
    float* out = (float*)d_out;

    const int N = in_sizes[0] / 8;               // B*L positions (C = 8)

    SSAccLoss_zero_ws<<<1, 32, 0, stream>>>(ws);
    SSAccLoss_accum<<<ACC_BLOCKS, ACC_THREADS, 0, stream>>>(real, pred, ws, N);
    SSAccLoss_finish<<<1, 32, 0, stream>>>(ws, out);
}
